// DimensionShuffler_34677565948082
// MI455X (gfx1250) — matrix operand order repaired and checked by reading, not compiled, not run on hardware
//
#include <hip/hip_runtime.h>

typedef __attribute__((ext_vector_type(16))) _Float16 v16h;
typedef __attribute__((ext_vector_type(8)))  _Float16 v8h;
typedef __attribute__((ext_vector_type(8)))  float    v8f;

#define MM 8192   // B*S
#define KK 1024   // IN
#define NN 4096   // OUT
#define LDST 40   // LDS row stride in halves (bank-conflict-free for 16B frag reads)

// ---------------- workspace layout (bytes) ----------------
// [0)        A f16  : MM*KK*2      = 16 MiB
// [A_end)    Bt f16 : NN*KK*2      =  8 MiB   (N-major: Bt[n][k])
// [Bt_end)   C f32  : MM*NN*4      = 128 MiB
// [C_end)    in_cs  : (KK/2)*2 f32 =  4 KiB   (cos,sin per in-pair)
// [incs_end) out_cs : (NN/2)*2 f32 = 16 KiB   (cos,sin per out-pair)
#define WS_A_OFF     0
#define WS_BT_OFF    ((size_t)MM*KK*2)
#define WS_C_OFF     (WS_BT_OFF + (size_t)NN*KK*2)
#define WS_INCS_OFF  (WS_C_OFF + (size_t)MM*NN*4)
#define WS_OUTCS_OFF (WS_INCS_OFF + (size_t)(KK/2)*2*4)

// ---------------- 1) trig tables ----------------
__global__ void k_trig(const float* __restrict__ in_sc, const float* __restrict__ out_sc,
                       float* __restrict__ incs, float* __restrict__ outcs) {
    int t = blockIdx.x * blockDim.x + threadIdx.x;
    if (t < KK / 2) {
        float a = in_sc[t];
        incs[2 * t]     = cosf(a);
        incs[2 * t + 1] = sinf(a);
    } else if (t < KK / 2 + NN / 2) {
        int i = t - KK / 2;
        float a = out_sc[i];
        outcs[2 * i]     = cosf(a);
        outcs[2 * i + 1] = sinf(a);
    }
}

// ---------------- 2) A = rotate(gather(input)) in f16 ----------------
__global__ void k_prepA(const float* __restrict__ x, const int* __restrict__ imap,
                        const float* __restrict__ incs, _Float16* __restrict__ A) {
    int t = blockIdx.x * blockDim.x + threadIdx.x;   // MM * (KK/2) threads
    int row = t >> 9;            // / (KK/2)
    int m   = t & (KK / 2 - 1);
    const float* xr = x + (size_t)row * KK;
    float x0 = xr[imap[2 * m]];
    float x1 = xr[imap[2 * m + 1]];
    float c = incs[2 * m], s = incs[2 * m + 1];
    union { _Float16 h[2]; unsigned u; } p;
    p.h[0] = (_Float16)(x0 * c + x1 * s);
    p.h[1] = (_Float16)(x1 * c - x0 * s);
    *(unsigned*)&A[(size_t)row * KK + 2 * m] = p.u;
}

// ---------------- 3) Bt[n][k] = mask.T * out_rot(in_rot(gather(W))) in f16 ----------------
__global__ void k_prepB(const float* __restrict__ W, const float* __restrict__ mask,
                        const int* __restrict__ imap, const int* __restrict__ omap,
                        const float* __restrict__ incs, const float* __restrict__ outcs,
                        _Float16* __restrict__ Bt) {
    int t = blockIdx.x * blockDim.x + threadIdx.x;   // (NN/2)*(KK/2) threads
    int p = t >> 9;              // out pair
    int m = t & (KK / 2 - 1);    // in pair
    int o0 = omap[2 * p], o1 = omap[2 * p + 1];
    int i0 = imap[2 * m], i1 = imap[2 * m + 1];
    float w00 = W[(size_t)o0 * KK + i0], w01 = W[(size_t)o0 * KK + i1];
    float w10 = W[(size_t)o1 * KK + i0], w11 = W[(size_t)o1 * KK + i1];
    float ci = incs[2 * m], si = incs[2 * m + 1];
    float a00 = w00 * ci + w01 * si, a01 = w01 * ci - w00 * si;
    float a10 = w10 * ci + w11 * si, a11 = w11 * ci - w10 * si;
    float co = outcs[2 * p], so = outcs[2 * p + 1];
    float b00 = a00 * co - a10 * so, b01 = a01 * co - a11 * so;   // out row 2p
    float b10 = a00 * so + a10 * co, b11 = a01 * so + a11 * co;   // out row 2p+1
    const float* mk0 = mask + (size_t)(2 * p) * KK + 2 * m;
    const float* mk1 = mk0 + KK;
    b00 *= mk0[0]; b01 *= mk0[1];
    b10 *= mk1[0]; b11 *= mk1[1];
    union { _Float16 h[2]; unsigned u; } q;
    q.h[0] = (_Float16)b00; q.h[1] = (_Float16)b01;
    *(unsigned*)&Bt[(size_t)(2 * p) * KK + 2 * m] = q.u;
    q.h[0] = (_Float16)b10; q.h[1] = (_Float16)b11;
    *(unsigned*)&Bt[(size_t)(2 * p + 1) * KK + 2 * m] = q.u;
}

// ---------------- 4) GEMM: C(MMxNN,f32) = A(MMxKK,f16) * Bt^T via WMMA ----------------
// Double-buffered LDS, filled with gfx1250 async global->LDS (ASYNCcnt) ops.
__global__ __launch_bounds__(256, 2)
void k_gemm(const _Float16* __restrict__ A, const _Float16* __restrict__ Bt,
            float* __restrict__ C) {
    __shared__ __align__(16) _Float16 As[2][128 * LDST];
    __shared__ __align__(16) _Float16 Bs[2][128 * LDST];
    const int tid  = threadIdx.x;
    const int lane = tid & 31, wave = tid >> 5;
    const int wm = wave & 1, wn = wave >> 1;          // 2x4 wave grid -> 128x128 block
    const int rowBlock = blockIdx.x * 128;
    const int colBlock = blockIdx.y * 128;
    const int lm = lane & 15, lh = lane >> 4;

    // this thread's two 16B staging chunks per 128x32 tile (512 chunks / 256 threads)
    const int r0 = tid >> 2,        c0 = tid & 3;
    const int r1 = (tid + 256) >> 2, c1 = (tid + 256) & 3;

    // issue async global->LDS loads for K-slab `kb` into buffer `buf`
    auto issue = [&](int buf, int kb) {
        unsigned aL0 = (unsigned)(unsigned long long)&As[buf][r0 * LDST + c0 * 8];
        unsigned aL1 = (unsigned)(unsigned long long)&As[buf][r1 * LDST + c1 * 8];
        unsigned bL0 = (unsigned)(unsigned long long)&Bs[buf][r0 * LDST + c0 * 8];
        unsigned bL1 = (unsigned)(unsigned long long)&Bs[buf][r1 * LDST + c1 * 8];
        unsigned long long aG0 =
            (unsigned long long)&A[(size_t)(rowBlock + r0) * KK + kb + c0 * 8];
        unsigned long long aG1 =
            (unsigned long long)&A[(size_t)(rowBlock + r1) * KK + kb + c1 * 8];
        unsigned long long bG0 =
            (unsigned long long)&Bt[(size_t)(colBlock + r0) * KK + kb + c0 * 8];
        unsigned long long bG1 =
            (unsigned long long)&Bt[(size_t)(colBlock + r1) * KK + kb + c1 * 8];
        asm volatile("global_load_async_to_lds_b128 %0, %1, off"
                     :: "v"(aL0), "v"(aG0) : "memory");
        asm volatile("global_load_async_to_lds_b128 %0, %1, off"
                     :: "v"(aL1), "v"(aG1) : "memory");
        asm volatile("global_load_async_to_lds_b128 %0, %1, off"
                     :: "v"(bL0), "v"(bG0) : "memory");
        asm volatile("global_load_async_to_lds_b128 %0, %1, off"
                     :: "v"(bL1), "v"(bG1) : "memory");
    };

    v8f acc[4][2] = {};

    issue(0, 0);                       // prologue: tile 0 -> buffer 0

    for (int kb = 0; kb < KK; kb += 32) {
        const int cur = (kb >> 5) & 1;
        // my async loads for this tile must have landed; barrier makes it
        // workgroup-wide (every wave waits its own ASYNCcnt before signaling).
        asm volatile("s_wait_asynccnt 0x0" ::: "memory");
        __syncthreads();
        if (kb + 32 < KK) {
            issue(cur ^ 1, kb + 32);   // overlap next-tile DMA with this tile's math
            if (kb + 64 < KK) {        // warm L2 for the slab after next
                __builtin_prefetch(&A[(size_t)(rowBlock + (tid >> 1)) * KK + kb + 64], 0, 0);
                __builtin_prefetch(&Bt[(size_t)(colBlock + (tid >> 1)) * KK + kb + 64], 0, 0);
            }
        }

        // A frag (16x32 f16): lane = M, lanes<16 hold K 0..7/16..23,
        // lanes>=16 hold K 8..15/24..31, packed 2 per VGPR.
        v16h af[4];
        for (int i = 0; i < 4; ++i) {
            const int r = wm * 64 + i * 16 + lm;
            v8h lo = *(const v8h*)&As[cur][r * LDST + lh * 8];
            v8h hi = *(const v8h*)&As[cur][r * LDST + 16 + lh * 8];
            for (int e = 0; e < 8; ++e) { af[i][e] = lo[e]; af[i][e + 8] = hi[e]; }
        }
        // B frag (32x16 f16): lane = N, lanes<16 hold K 0..7/16..23, lanes>=16 hold K 8..15/24..31.
        v16h bf[2];
        for (int j = 0; j < 2; ++j) {
            const int n = wn * 32 + j * 16 + lm;
            v8h lo = *(const v8h*)&Bs[cur][n * LDST + lh * 8];
            v8h hi = *(const v8h*)&Bs[cur][n * LDST + 16 + lh * 8];
            for (int e = 0; e < 8; ++e) { bf[j][e] = lo[e]; bf[j][e + 8] = hi[e]; }
        }
        for (int i = 0; i < 4; ++i)
            for (int j = 0; j < 2; ++j)
                acc[i][j] = __builtin_amdgcn_wmma_f32_16x16x32_f16(
                    false, af[i], false, bf[j], (short)0, acc[i][j], false, false);
    }

    // C layout: VGPR r holds row = r + (lane>=16)*8, col = lane%16
    for (int i = 0; i < 4; ++i)
        for (int j = 0; j < 2; ++j) {
            const int col   = colBlock + wn * 32 + j * 16 + lm;
            const int rbase = rowBlock + wm * 64 + i * 16 + lh * 8;
            for (int r = 0; r < 8; ++r)
                C[(size_t)(rbase + r) * NN + col] = acc[i][j][r];
        }
}

// ---------------- 5) epilogue: out[:,j] = out_rot(C pair)[omr[j]] ----------------
__global__ void k_epilogue(const float* __restrict__ C, const int* __restrict__ omr,
                           const float* __restrict__ outcs, float* __restrict__ out) {
    long long t = (long long)blockIdx.x * blockDim.x + threadIdx.x;  // MM*NN
    int j = (int)(t & (NN - 1));
    size_t row = (size_t)(t >> 12);
    int idx = omr[j];
    int p = idx >> 1;
    float co = outcs[2 * p], so = outcs[2 * p + 1];
    const float* cr = C + row * NN + 2 * p;
    float y0 = cr[0], y1 = cr[1];
    float z = (idx & 1) ? (y1 * co - y0 * so) : (y0 * co + y1 * so);
    out[row * NN + j] = z;
}

extern "C" void kernel_launch(void* const* d_in, const int* in_sizes, int n_in,
                              void* d_out, int out_size, void* d_ws, size_t ws_size,
                              hipStream_t stream) {
    const float* input  = (const float*)d_in[0];
    const float* weight = (const float*)d_in[1];
    const float* mask   = (const float*)d_in[2];
    const float* in_sc  = (const float*)d_in[3];
    const float* out_sc = (const float*)d_in[4];
    const int*   imap   = (const int*)d_in[5];
    const int*   omap   = (const int*)d_in[6];
    const int*   omr    = (const int*)d_in[7];
    float* out = (float*)d_out;

    char* ws = (char*)d_ws;
    _Float16* A  = (_Float16*)(ws + WS_A_OFF);
    _Float16* Bt = (_Float16*)(ws + WS_BT_OFF);
    float* C     = (float*)(ws + WS_C_OFF);
    float* incs  = (float*)(ws + WS_INCS_OFF);
    float* outcs = (float*)(ws + WS_OUTCS_OFF);

    k_trig<<<(KK / 2 + NN / 2 + 255) / 256, 256, 0, stream>>>(in_sc, out_sc, incs, outcs);
    k_prepA<<<(MM * (KK / 2)) / 256, 256, 0, stream>>>(input, imap, incs, A);
    k_prepB<<<((NN / 2) * (KK / 2)) / 256, 256, 0, stream>>>(weight, mask, imap, omap,
                                                             incs, outcs, Bt);
    dim3 ggrid(MM / 128, NN / 128);
    k_gemm<<<ggrid, 256, 0, stream>>>(A, Bt, C);
    k_epilogue<<<(int)(((long long)MM * NN) / 256), 256, 0, stream>>>(C, omr, outcs, out);
}
